// PointNet2_7473243095566
// MI455X (gfx1250) — matrix operand order repaired and checked by reading, not compiled, not run on hardware
//
#include <hip/hip_runtime.h>
#include <hip/hip_bf16.h>

typedef __bf16 bf16;
typedef __attribute__((ext_vector_type(16))) __bf16 v16bf;
typedef __attribute__((ext_vector_type(8)))  float  v8f;

#define BB 4
#define USE_ASYNC_LDS 1

// ---------------------------------------------------------------------------
// CDNA5 async global->LDS copy (ASYNCcnt path, ISA 08_async_tensor §4)
// ---------------------------------------------------------------------------
__device__ __forceinline__ void async_copy16(unsigned ldsoff, const void* g) {
#if USE_ASYNC_LDS
    asm volatile("global_load_async_to_lds_b128 %0, %1, off"
                 :: "v"(ldsoff), "v"((unsigned long long)g) : "memory");
#else
    (void)ldsoff; (void)g;
#endif
}
__device__ __forceinline__ void async_wait_all() {
#if USE_ASYNC_LDS
    asm volatile("s_wait_asynccnt 0" ::: "memory");
#endif
}

// ---------------------------------------------------------------------------
// Weight pack: f32 (M x Cin) -> bf16 (M x Kpad), zero padded
// ---------------------------------------------------------------------------
__global__ void pack_w_kernel(const float* __restrict__ W, bf16* __restrict__ dst,
                              int M, int Cin, int Kpad) {
    int i = blockIdx.x * blockDim.x + threadIdx.x;
    if (i >= M * Kpad) return;
    int m = i / Kpad, k = i % Kpad;
    dst[i] = (k < Cin) ? (bf16)W[(size_t)m * Cin + k] : (bf16)0.0f;
}

// ---------------------------------------------------------------------------
// Init: split pointcloud -> xyz0 (ws + d_out region0) and act0 (bf16, 32ch pad)
// ---------------------------------------------------------------------------
__global__ void init_pts_kernel(const float* __restrict__ pc, float* __restrict__ xyz0,
                                float* __restrict__ out_xyz, bf16* __restrict__ act0, int N) {
    int i = blockIdx.x * blockDim.x + threadIdx.x;
    if (i >= BB * N) return;
    const float* p = pc + (size_t)i * 6;
    float x = p[0], y = p[1], z = p[2];
    xyz0[i * 3 + 0] = x; xyz0[i * 3 + 1] = y; xyz0[i * 3 + 2] = z;
    out_xyz[i * 3 + 0] = x; out_xyz[i * 3 + 1] = y; out_xyz[i * 3 + 2] = z;
    bf16* r = act0 + (size_t)i * 32;
    r[0] = (bf16)p[3]; r[1] = (bf16)p[4]; r[2] = (bf16)p[5];
    for (int c = 3; c < 32; ++c) r[c] = (bf16)0.0f;
}

// ---------------------------------------------------------------------------
// Farthest point sampling: one block per batch, LDS distance array
// ---------------------------------------------------------------------------
__global__ void fps_kernel(const float* __restrict__ xyz, int N, int npoint,
                           int* __restrict__ out_idx) {
    __shared__ float dist[8192];
    __shared__ float rv[256];
    __shared__ int   ri[256];
    int b = blockIdx.x, tid = threadIdx.x, nt = blockDim.x;
    const float* X = xyz + (size_t)b * N * 3;
    for (int i = tid; i < N; i += nt) dist[i] = 1e10f;
    __syncthreads();
    int far = 0;
    for (int it = 0; it < npoint; ++it) {
        if (tid == 0) out_idx[(size_t)b * npoint + it] = far;
        float fx = X[far * 3 + 0], fy = X[far * 3 + 1], fz = X[far * 3 + 2];
        float bv = -1.0f; int bi = 0;
        for (int i = tid; i < N; i += nt) {
            float dx = X[i * 3 + 0] - fx, dy = X[i * 3 + 1] - fy, dz = X[i * 3 + 2] - fz;
            float d = dx * dx + dy * dy + dz * dz;
            float nd = fminf(dist[i], d);
            dist[i] = nd;
            if (nd > bv) { bv = nd; bi = i; }
        }
        rv[tid] = bv; ri[tid] = bi;
        __syncthreads();
        for (int s = nt >> 1; s > 0; s >>= 1) {
            if (tid < s) {
                if (rv[tid + s] > rv[tid] ||
                    (rv[tid + s] == rv[tid] && ri[tid + s] < ri[tid])) {
                    rv[tid] = rv[tid + s]; ri[tid] = ri[tid + s];
                }
            }
            __syncthreads();
        }
        far = ri[0];
        __syncthreads();
    }
}

// ---------------------------------------------------------------------------
// Gather sampled centers; optional second destination (mid_xyz output)
// ---------------------------------------------------------------------------
__global__ void gather_xyz_kernel(const float* __restrict__ xyz, const int* __restrict__ fidx,
                                  float* __restrict__ nx, float* __restrict__ nx2,
                                  int N, int np) {
    int i = blockIdx.x * blockDim.x + threadIdx.x;
    if (i >= BB * np) return;
    int b = i / np;
    int p = fidx[i];
    const float* s = xyz + ((size_t)b * N + p) * 3;
    float x = s[0], y = s[1], z = s[2];
    nx[i * 3 + 0] = x; nx[i * 3 + 1] = y; nx[i * 3 + 2] = z;
    if (nx2) { nx2[i * 3 + 0] = x; nx2[i * 3 + 1] = y; nx2[i * 3 + 2] = z; }
}

// ---------------------------------------------------------------------------
// Ball query: first 32 indices (ascending) within radius, pad with first
// ---------------------------------------------------------------------------
__global__ void ball_query_kernel(const float* __restrict__ xyz, const float* __restrict__ nxyz,
                                  int* __restrict__ idx, int N, int np, float r2) {
    int i = blockIdx.x * blockDim.x + threadIdx.x;
    if (i >= BB * np) return;
    int b = i / np;
    const float* X = xyz + (size_t)b * N * 3;
    float cx = nxyz[i * 3 + 0], cy = nxyz[i * 3 + 1], cz = nxyz[i * 3 + 2];
    int* o = idx + (size_t)i * 32;
    int cnt = 0, first = -1;
    for (int n = 0; n < N && cnt < 32; ++n) {
        float dx = X[n * 3 + 0] - cx, dy = X[n * 3 + 1] - cy, dz = X[n * 3 + 2] - cz;
        float d = dx * dx + dy * dy + dz * dz;
        if (d < r2) { if (first < 0) first = n; o[cnt++] = n; }
    }
    if (first < 0) first = N - 1;
    for (; cnt < 32; ++cnt) o[cnt] = first;
}

// ---------------------------------------------------------------------------
// Grouping: build bf16 activation rows [xyz_rel(3) | prev feats(C) | 0 pad]
// ---------------------------------------------------------------------------
__global__ void group_kernel(const float* __restrict__ xyz, const float* __restrict__ nxyz,
                             const int* __restrict__ idx, const bf16* __restrict__ actp,
                             int rowPrev, int C, bf16* __restrict__ g, int Kpad,
                             int N, int np) {
    int i = blockIdx.x * blockDim.x + threadIdx.x;
    if (i >= BB * np * 32) return;
    int s = (i >> 5) % np;
    int b = (i >> 5) / np;
    int p = idx[i];
    const float* X  = xyz  + ((size_t)b * N  + p) * 3;
    const float* cc = nxyz + ((size_t)b * np + s) * 3;
    bf16* o = g + (size_t)i * Kpad;
    o[0] = (bf16)(X[0] - cc[0]);
    o[1] = (bf16)(X[1] - cc[1]);
    o[2] = (bf16)(X[2] - cc[2]);
    const bf16* f = actp + ((size_t)b * N + p) * rowPrev;
    int c = 0;
    for (; c < C; ++c) o[3 + c] = f[c];
    for (; 3 + c < Kpad; ++c) o[3 + c] = (bf16)0.0f;
}

// ---------------------------------------------------------------------------
// Max pool over 32 samples per center
// ---------------------------------------------------------------------------
__global__ void maxpool_kernel(const bf16* __restrict__ in, bf16* __restrict__ out,
                               int np, int C) {
    int i = blockIdx.x * blockDim.x + threadIdx.x;
    if (i >= BB * np * C) return;
    int c = i % C;
    int s = (i / C) % np;
    int b = i / (C * np);
    const bf16* base = in + (((size_t)b * np + s) * 32) * C + c;
    float m = -1e30f;
    for (int j = 0; j < 32; ++j) m = fmaxf(m, (float)base[(size_t)j * C]);
    out[i] = (bf16)m;
}

// ---------------------------------------------------------------------------
// FP: 3-NN inverse-distance interp + concat unknown feats -> bf16 rows
// ---------------------------------------------------------------------------
__global__ void fp_interp_kernel(const float* __restrict__ uxyz, const float* __restrict__ kxyz,
                                 const bf16* __restrict__ kfeat, int Ck, int kRow,
                                 const bf16* __restrict__ ufeat, int Cu, int uRow,
                                 bf16* __restrict__ g, int Kpad, int Nu, int Nk) {
    int i = blockIdx.x * blockDim.x + threadIdx.x;
    if (i >= BB * Nu) return;
    int b = i / Nu;
    const float* U = uxyz + (size_t)i * 3;
    const float* K = kxyz + (size_t)b * Nk * 3;
    float ux = U[0], uy = U[1], uz = U[2];
    float d0 = 1e30f, d1 = 1e30f, d2v = 1e30f;
    int i0 = 0, i1 = 0, i2 = 0;
    for (int k = 0; k < Nk; ++k) {
        float dx = K[k * 3 + 0] - ux, dy = K[k * 3 + 1] - uy, dz = K[k * 3 + 2] - uz;
        float d = dx * dx + dy * dy + dz * dz;
        if (d < d0)      { d2v = d1; i2 = i1; d1 = d0; i1 = i0; d0 = d; i0 = k; }
        else if (d < d1) { d2v = d1; i2 = i1; d1 = d;  i1 = k; }
        else if (d < d2v){ d2v = d;  i2 = k; }
    }
    float w0 = 1.0f / (d0 + 1e-8f), w1 = 1.0f / (d1 + 1e-8f), w2 = 1.0f / (d2v + 1e-8f);
    float ws = w0 + w1 + w2; w0 /= ws; w1 /= ws; w2 /= ws;
    const bf16* f0 = kfeat + ((size_t)b * Nk + i0) * kRow;
    const bf16* f1 = kfeat + ((size_t)b * Nk + i1) * kRow;
    const bf16* f2 = kfeat + ((size_t)b * Nk + i2) * kRow;
    bf16* o = g + (size_t)i * Kpad;
    for (int c = 0; c < Ck; ++c)
        o[c] = (bf16)(w0 * (float)f0[c] + w1 * (float)f1[c] + w2 * (float)f2[c]);
    const bf16* uf = ufeat + (size_t)i * uRow;
    for (int c = 0; c < Cu; ++c) o[Ck + c] = uf[c];
    for (int c = Ck + Cu; c < Kpad; ++c) o[c] = (bf16)0.0f;
}

// ---------------------------------------------------------------------------
// GEMM + BN + ReLU via v_wmma_f32_16x16x32_bf16, LDS-staged + async loads.
//
// act: (B, Nrows, Kpad) bf16 point-major; Wp: (M, Kpad) bf16.
// Block: 256 threads = 8 waves; covers full M (mw = M/64 wave-rows) and
// N_BLK = (8/mw)*32 columns. Each wave computes a 64x32 tile = 4x2 WMMA accs.
// Per K-step a (M + N_BLK) x 32 bf16 slab is staged into LDS (double
// buffered) with global_load_async_to_lds_b128 and consumed via ds loads.
// outAct: (B, Nrows, M) bf16   |   outF32: (B, M, Nrows) f32 channel-major
// ---------------------------------------------------------------------------
__global__ void __launch_bounds__(256)
gemm_bn_relu_kernel(const bf16* __restrict__ act, const bf16* __restrict__ Wp,
                    const float* __restrict__ gamma, const float* __restrict__ beta,
                    int M, int Kpad, int Nrows, int mw,
                    bf16* __restrict__ outAct, float* __restrict__ outF32, int doRelu) {
    __shared__ __align__(16) bf16 smem[2][10240];   // 2 x 20KB (<= (256+64)*32)

    int b    = blockIdx.z;
    int tid  = threadIdx.x;
    int wave = tid >> 5;
    int lane = tid & 31;
    int nw    = 8 / mw;
    int M_BLK = mw * 64;            // == M for all layers here
    int N_BLK = nw * 32;
    int n_blk0 = blockIdx.x * N_BLK;

    const bf16* Abase = act + (size_t)b * Nrows * Kpad;

    int mwav = wave % mw, nwav = wave / mw;
    int m0w = mwav * 64;            // 4 m-subtiles
    int n0w = nwav * 32;            // 2 n-subtiles
    int khalf = lane >> 4;          // ISA 7.12.2 lane halves
    int lr    = lane & 15;

    int totalRows = M_BLK + N_BLK;  // 64B (32 bf16) per row per K-step
    int chunks    = totalRows * 4;  // 16B chunks

    v8f acc[4][2];
    #pragma unroll
    for (int i = 0; i < 4; ++i)
        #pragma unroll
        for (int j = 0; j < 2; ++j)
            acc[i][j] = v8f{0.f,0.f,0.f,0.f,0.f,0.f,0.f,0.f};

    // Stage one K-slab into LDS buffer `buf` (W rows first, then act rows).
    auto stage = [&](int buf, int k0) {
        for (int c = tid; c < chunks; c += 256) {
            int row = c >> 2, part = c & 3;
            const bf16* src = (row < M_BLK)
                ? (Wp   + (size_t)row * Kpad + k0)
                : (Abase + (size_t)(n_blk0 + row - M_BLK) * Kpad + k0);
            const char* gp = (const char*)src + part * 16;
            bf16* dp = &smem[buf][row * 32 + part * 8];
#if USE_ASYNC_LDS
            async_copy16((unsigned)(size_t)dp, gp);
#else
            *(int4*)dp = *(const int4*)gp;
#endif
        }
    };

    stage(0, 0);
    async_wait_all();
    __syncthreads();

    int cur = 0;
    for (int k0 = 0; k0 < Kpad; k0 += 32) {
        int nxt = cur ^ 1;
        bool more = (k0 + 32) < Kpad;
        if (more) stage(nxt, k0 + 32);

        const bf16* Wt = &smem[cur][0];              // M_BLK x 32
        const bf16* At = &smem[cur][M_BLK * 32];     // N_BLK x 32

        v16bf avf[4], bvf[2];
        #pragma unroll
        for (int i = 0; i < 4; ++i) {
            const bf16* wr = Wt + (size_t)(m0w + i * 16 + lr) * 32;
            #pragma unroll
            for (int e = 0; e < 8; ++e) {
                avf[i][e]     = wr[khalf * 8 + e];
                avf[i][e + 8] = wr[16 + khalf * 8 + e];
            }
        }
        #pragma unroll
        for (int j = 0; j < 2; ++j) {
            const bf16* br = At + (size_t)(n0w + j * 16 + lr) * 32;
            #pragma unroll
            for (int e = 0; e < 8; ++e) {
                bvf[j][e]     = br[khalf * 8 + e];
                bvf[j][e + 8] = br[16 + khalf * 8 + e];
            }
        }
        #pragma unroll
        for (int i = 0; i < 4; ++i)
            #pragma unroll
            for (int j = 0; j < 2; ++j)
                acc[i][j] = __builtin_amdgcn_wmma_f32_16x16x32_bf16(
                    false, avf[i], false, bvf[j], (short)0, acc[i][j], false, false);

        if (more) async_wait_all();
        __syncthreads();
        cur = nxt;
    }

    // Epilogue: BN + ReLU, write bf16 activation and/or f32 channel-major.
    #pragma unroll
    for (int j = 0; j < 2; ++j) {
        int n = n_blk0 + n0w + j * 16 + lr;
        #pragma unroll
        for (int i = 0; i < 4; ++i) {
            #pragma unroll
            for (int v = 0; v < 8; ++v) {
                int m = m0w + i * 16 + khalf * 8 + v;
                float val = acc[i][j][v];
                if (gamma)      val = gamma[m] * val + beta[m];
                else if (beta)  val = val + beta[m];
                if (doRelu)     val = fmaxf(val, 0.0f);
                if (outAct) outAct[((size_t)b * Nrows + n) * M + m] = (bf16)val;
                if (outF32) outF32[((size_t)b * M + m) * Nrows + n] = val;
            }
        }
    }
}

// ---------------------------------------------------------------------------
// Host orchestration
// ---------------------------------------------------------------------------
static inline int ceil32(int x) { return (x + 31) & ~31; }

extern "C" void kernel_launch(void* const* d_in, const int* in_sizes, int n_in,
                              void* d_out, int out_size, void* d_ws, size_t ws_size,
                              hipStream_t stream) {
    (void)in_sizes; (void)n_in; (void)out_size; (void)ws_size;
    const int N = 8192, NP1 = 2048, NP2 = 512, NP3 = 128;
    const float R2[3] = {0.09f, 0.25f, 0.49f};

    // ---- workspace bump allocator (256B aligned) ----
    char* ws = (char*)d_ws;
    size_t off = 0;
    auto alloc = [&](size_t bytes) -> void* {
        void* p = ws + off;
        off = (off + bytes + 255) & ~(size_t)255;
        return p;
    };

    float* xyz0 = (float*)alloc((size_t)BB * N   * 3 * 4);
    float* xyz1 = (float*)alloc((size_t)BB * NP1 * 3 * 4);
    float* xyz2 = (float*)alloc((size_t)BB * NP2 * 3 * 4);
    float* xyz3 = (float*)alloc((size_t)BB * NP3 * 3 * 4);
    bf16* act0      = (bf16*)alloc((size_t)BB * N   * 32  * 2); // 3ch pad->32
    bf16* act_feat1 = (bf16*)alloc((size_t)BB * NP1 * 128 * 2);
    bf16* act_feat2 = (bf16*)alloc((size_t)BB * NP2 * 256 * 2);
    bf16* act_feat3 = (bf16*)alloc((size_t)BB * NP3 * 256 * 2);
    bf16* fp_feat2  = (bf16*)alloc((size_t)BB * NP2 * 128 * 2);
    bf16* fp_feat1  = (bf16*)alloc((size_t)BB * NP1 * 256 * 2);
    int* fidx = (int*)alloc((size_t)BB * NP1 * 4);            // reused per level
    int* bidx = (int*)alloc((size_t)BB * NP1 * 32 * 4);       // reused per level
    bf16* bufG = (bf16*)alloc((size_t)24 * 1024 * 1024);      // grouped/interp rows
    bf16* bufA = (bf16*)alloc((size_t)64 * 1024 * 1024);      // layer ping
    bf16* bufB = (bf16*)alloc((size_t)32 * 1024 * 1024);      // layer pong

    // ---- pack all weights to bf16 padded ----
    struct Layer { const float* W; const float* g; const float* bta; int M, Cin, Kpad; bf16* P; };
    Layer L[16];
    int nl = 0;
    const int saD[3][4] = {{6,64,64,128},{131,128,128,256},{259,256,256,256}};
    const int fpD[3][3] = {{512,128,128},{256,256,256},{259,256,256}};
    for (int g = 0; g < 3; ++g)
        for (int l = 0; l < 3; ++l) {
            int base = 1 + g * 9 + l * 3;
            L[nl++] = { (const float*)d_in[base], (const float*)d_in[base+1],
                        (const float*)d_in[base+2], saD[g][l+1], saD[g][l],
                        ceil32(saD[g][l]), nullptr };
        }
    for (int g = 0; g < 3; ++g)
        for (int l = 0; l < 2; ++l) {
            int base = 28 + g * 6 + l * 3;
            L[nl++] = { (const float*)d_in[base], (const float*)d_in[base+1],
                        (const float*)d_in[base+2], fpD[g][l+1], fpD[g][l],
                        ceil32(fpD[g][l]), nullptr };
        }
    L[nl++] = { (const float*)d_in[46], nullptr, (const float*)d_in[47], 256, 256, 256, nullptr };
    for (int i = 0; i < 16; ++i) {
        L[i].P = (bf16*)alloc((size_t)L[i].M * L[i].Kpad * 2);
        int tot = L[i].M * L[i].Kpad;
        pack_w_kernel<<<dim3((tot + 255) / 256), 256, 0, stream>>>(
            L[i].W, L[i].P, L[i].M, L[i].Cin, L[i].Kpad);
    }

    auto gemm = [&](const bf16* act, const Layer& ly, int Nrows,
                    bf16* oAct, float* oF32, int relu) {
        int mw = ly.M / 64;                    // 1, 2 or 4
        int N_BLK = (8 / mw) * 32;             // 256, 128 or 64
        dim3 grid(Nrows / N_BLK, 1, BB);
        gemm_bn_relu_kernel<<<grid, 256, 0, stream>>>(
            act, ly.P, ly.g, ly.bta, ly.M, ly.Kpad, Nrows, mw, oAct, oF32, relu);
    };

    // ---- d_out regions (fp32) ----
    float* out_xyz0 = (float*)d_out;                                      // (4,8192,3)
    float* out_feat = out_xyz0 + (size_t)BB * N * 3;                      // (4,256,8192)
    float* out_midx = out_feat + (size_t)BB * 256 * N;                    // (4,128,3)
    float* out_midf = out_midx + (size_t)BB * NP3 * 3;                    // (4,256,128)

    // ---- init ----
    init_pts_kernel<<<dim3((BB * N + 255) / 256), 256, 0, stream>>>(
        (const float*)d_in[0], xyz0, out_xyz0, act0, N);

    // ---- SA levels ----
    struct SAL { const float* xyz; float* nxyz; float* nxyz2; int Ni, np;
                 const bf16* actp; int rowPrev, C; int li; bf16* pooled; };
    SAL sa[3] = {
        { xyz0, xyz1, nullptr,  N,   NP1, act0,      32,  3,   0, act_feat1 },
        { xyz1, xyz2, nullptr,  NP1, NP2, act_feat1, 128, 128, 3, act_feat2 },
        { xyz2, xyz3, out_midx, NP2, NP3, act_feat2, 256, 256, 6, act_feat3 },
    };
    for (int lv = 0; lv < 3; ++lv) {
        SAL& s = sa[lv];
        fps_kernel<<<dim3(BB), 256, 0, stream>>>(s.xyz, s.Ni, s.np, fidx);
        gather_xyz_kernel<<<dim3((BB * s.np + 255) / 256), 256, 0, stream>>>(
            s.xyz, fidx, s.nxyz, s.nxyz2, s.Ni, s.np);
        ball_query_kernel<<<dim3((BB * s.np + 255) / 256), 256, 0, stream>>>(
            s.xyz, s.nxyz, bidx, s.Ni, s.np, R2[lv]);
        int Kp = L[s.li].Kpad;                 // ceil32(3 + C)
        int rows = s.np * 32;                  // per batch
        group_kernel<<<dim3((BB * rows + 255) / 256), 256, 0, stream>>>(
            s.xyz, s.nxyz, bidx, s.actp, s.rowPrev, s.C, bufG, Kp, s.Ni, s.np);
        gemm(bufG, L[s.li + 0], rows, bufA, nullptr, 1);
        gemm(bufA, L[s.li + 1], rows, bufB, nullptr, 1);
        gemm(bufB, L[s.li + 2], rows, bufA, nullptr, 1);
        int Cl = L[s.li + 2].M;
        maxpool_kernel<<<dim3((BB * s.np * Cl + 255) / 256), 256, 0, stream>>>(
            bufA, s.pooled, s.np, Cl);
    }

    // ---- transition: mid_feat = trans_W @ feat3 + trans_b (no relu) ----
    gemm(act_feat3, L[15], NP3, nullptr, out_midf, 0);

    // ---- FP1: unknown=level2 (512), known=level3 (128) ----
    fp_interp_kernel<<<dim3((BB * NP2 + 255) / 256), 256, 0, stream>>>(
        xyz2, xyz3, act_feat3, 256, 256, act_feat2, 256, 256, bufG, 512, NP2, NP3);
    gemm(bufG, L[9],  NP2, bufA, nullptr, 1);
    gemm(bufA, L[10], NP2, fp_feat2, nullptr, 1);

    // ---- FP2: unknown=level1 (2048), known=level2 (512) ----
    fp_interp_kernel<<<dim3((BB * NP1 + 255) / 256), 256, 0, stream>>>(
        xyz1, xyz2, fp_feat2, 128, 128, act_feat1, 128, 128, bufG, 256, NP1, NP2);
    gemm(bufG, L[11], NP1, bufA, nullptr, 1);
    gemm(bufA, L[12], NP1, fp_feat1, nullptr, 1);

    // ---- FP3: unknown=level0 (8192), known=level1 (2048); final -> f32 out ----
    fp_interp_kernel<<<dim3((BB * N + 255) / 256), 256, 0, stream>>>(
        xyz0, xyz1, fp_feat1, 256, 256, act0, 3, 32, bufG, 288, N, NP1);
    gemm(bufG, L[13], N, bufA, nullptr, 1);
    gemm(bufA, L[14], N, nullptr, out_feat, 1);
}
